// RGLRU_65687229825730
// MI455X (gfx1250) — matrix operand order repaired and checked by reading, not compiled, not run on hardware
//
#include <hip/hip_runtime.h>

// ---------------------------------------------------------------------------
// RG-LRU for MI455X (gfx1250, wave32):
//   Phase 1: fused dual-gate GEMM (x@W_r^T, x@W_i^T) via v_wmma_f32_16x16x32_bf16
//            with fused sigmoid/exp epilogue producing a[] and gx[].
//            Wave tile 32x32 per gate -> 8 WMMAs per 6 LDS fragments.
//   Phase 2: chunked linear scan h = a*h + gx (3 passes, massively parallel).
// ---------------------------------------------------------------------------

typedef __attribute__((ext_vector_type(16))) __bf16 v16bf;
typedef __attribute__((ext_vector_type(8)))  float  v8f;

struct alignas(16) U128 { unsigned int x, y, z, w; };
struct alignas(32) U256 { U128 lo, hi; };

// Pack two fp32 into packed bf16 (RNE) using only 32-bit integer ops.
static __device__ __forceinline__ unsigned pack_bf16x2(float lo, float hi) {
  unsigned ul = __float_as_uint(lo);
  unsigned uh = __float_as_uint(hi);
  ul += 0x7FFFu + ((ul >> 16) & 1u);
  uh += 0x7FFFu + ((uh >> 16) & 1u);
  return (ul >> 16) | (uh & 0xFFFF0000u);
}
static __device__ __forceinline__ float sigmoidf_(float z) {
  return 1.0f / (1.0f + __expf(-z));
}
static __device__ __forceinline__ float softplusf_(float z) {
  return fmaxf(z, 0.0f) + log1pf(__expf(-fabsf(z)));
}

// Problem sizes (fixed by the reference).
constexpr int Bsz = 4, T = 4096, D = 1024;
constexpr int M   = Bsz * T;          // 16384 rows
constexpr int Kd  = D;                // reduction depth
constexpr int BM = 64, BN = 128, BK = 64;
constexpr int LDA = 88;               // padded LDS row stride (ushorts): 176B = 44 banks,
constexpr int LDB = 88;               // conflict-free across 16 rows, 16B-aligned for b128
constexpr int CH = 64, LT = T / CH;   // scan: 64 chunks of 64 steps

static __device__ __forceinline__ v16bf load_frag(const unsigned short* lds, int off,
                                                  int stride_u) {
  U256 p;
  p.lo = *(const U128*)&lds[off];
  p.hi = *(const U128*)&lds[off + stride_u];
  return __builtin_bit_cast(v16bf, p);
}

// ---------------------------------------------------------------------------
// Phase 1: dual-gate GEMM + gating epilogue.
// Workgroup tile 64(M) x 128(N); 8 waves arranged 2(M) x 4(N); each wave owns
// a 32x32 output tile for BOTH gates (8 accumulators). K consumed 64/step.
// ---------------------------------------------------------------------------
__global__ __launch_bounds__(256)
void rglru_gates_wmma(const float* __restrict__ x,
                      const float* __restrict__ W_r, const float* __restrict__ b_r,
                      const float* __restrict__ W_i, const float* __restrict__ b_i,
                      const float* __restrict__ Lambda,
                      float* __restrict__ a_out, float* __restrict__ gx_out)
{
  __shared__ unsigned short ldsA [BM * LDA];   // 11.0 KB
  __shared__ unsigned short ldsBr[BN * LDB];   // 22.0 KB
  __shared__ unsigned short ldsBi[BN * LDB];   // 22.0 KB

  const int tid  = threadIdx.x;
  const int lane = tid & 31;
  const int wv   = tid >> 5;
  const int wvM  = wv & 1;            // 2 waves in M -> 32-row slice
  const int wvN  = wv >> 1;           // 4 waves in N -> 32-col slice
  const int m0   = blockIdx.y * BM;
  const int n0   = blockIdx.x * BN;
  const int lr   = lane & 15;
  const int hi   = lane >> 4;

  // acc[gate][mi][ni]
  v8f accR00 = {}, accR01 = {}, accR10 = {}, accR11 = {};
  v8f accI00 = {}, accI01 = {}, accI10 = {}, accI11 = {};

  for (int k0 = 0; k0 < Kd; k0 += BK) {
    __syncthreads();
    // --- Stage A tile: 64x64 fp32 -> bf16 (row-major, padded) --------------
    #pragma unroll
    for (int i = 0; i < 4; ++i) {
      const int v   = tid + i * 256;        // 1024 float4 loads
      const int row = v >> 4;               // 16 float4 per row
      const int c4  = (v & 15) << 2;
      const float4 f = *(const float4*)(x + (size_t)(m0 + row) * Kd + k0 + c4);
      uint2 pk;
      pk.x = pack_bf16x2(f.x, f.y);
      pk.y = pack_bf16x2(f.z, f.w);
      *(uint2*)&ldsA[row * LDA + c4] = pk;
    }
    // --- Stage B tiles: W row-major over K == native WMMA B order ----------
    #pragma unroll
    for (int i = 0; i < 8; ++i) {
      const int v   = tid + i * 256;        // 2048 float4 loads per gate
      const int row = v >> 4;               // 0..127
      const int c4  = (v & 15) << 2;
      const size_t wo = (size_t)(n0 + row) * Kd + k0 + c4;
      const float4 fr = *(const float4*)(W_r + wo);
      const float4 fi = *(const float4*)(W_i + wo);
      uint2 pr, pi;
      pr.x = pack_bf16x2(fr.x, fr.y);
      pr.y = pack_bf16x2(fr.z, fr.w);
      pi.x = pack_bf16x2(fi.x, fi.y);
      pi.y = pack_bf16x2(fi.z, fi.w);
      *(uint2*)&ldsBr[row * LDB + c4] = pr;
      *(uint2*)&ldsBi[row * LDB + c4] = pi;
    }
    // Prefetch next K slab of x into cache (global_prefetch_b8).
    if (k0 + BK < Kd) {
      __builtin_prefetch(x + (size_t)(m0 + (tid >> 2)) * Kd + k0 + BK + ((tid & 3) << 4), 0, 0);
    }
    __syncthreads();

    // --- Compute: 2 K-slabs of 32; per slab 8 WMMAs from 6 fragments -------
    #pragma unroll
    for (int kb = 0; kb < 2; ++kb) {
      // A fragments (16x32 bf16): lane<16 -> K[0..7],K[16..23]; lane>=16 -> K[8..15],K[24..31]
      const int aoff0 = (wvM * 32 + lr)      * LDA + kb * 32 + hi * 8;
      const int aoff1 = (wvM * 32 + 16 + lr) * LDA + kb * 32 + hi * 8;
      const v16bf a0 = load_frag(ldsA, aoff0, 16);
      const v16bf a1 = load_frag(ldsA, aoff1, 16);
      // B fragments (32x16 bf16): lane<16 -> K[0..7],K[16..23] of col n; lane>=16 -> K[8..15],K[24..31]
      const int boff0 = (wvN * 32 + lr)      * LDB + kb * 32 + hi * 8;
      const int boff1 = (wvN * 32 + 16 + lr) * LDB + kb * 32 + hi * 8;
      const v16bf br0 = load_frag(ldsBr, boff0, 16);
      const v16bf br1 = load_frag(ldsBr, boff1, 16);
      const v16bf bi0 = load_frag(ldsBi, boff0, 16);
      const v16bf bi1 = load_frag(ldsBi, boff1, 16);

      accR00 = __builtin_amdgcn_wmma_f32_16x16x32_bf16(false, a0, false, br0, (short)0, accR00, false, false);
      accR01 = __builtin_amdgcn_wmma_f32_16x16x32_bf16(false, a0, false, br1, (short)0, accR01, false, false);
      accR10 = __builtin_amdgcn_wmma_f32_16x16x32_bf16(false, a1, false, br0, (short)0, accR10, false, false);
      accR11 = __builtin_amdgcn_wmma_f32_16x16x32_bf16(false, a1, false, br1, (short)0, accR11, false, false);
      accI00 = __builtin_amdgcn_wmma_f32_16x16x32_bf16(false, a0, false, bi0, (short)0, accI00, false, false);
      accI01 = __builtin_amdgcn_wmma_f32_16x16x32_bf16(false, a0, false, bi1, (short)0, accI01, false, false);
      accI10 = __builtin_amdgcn_wmma_f32_16x16x32_bf16(false, a1, false, bi0, (short)0, accI10, false, false);
      accI11 = __builtin_amdgcn_wmma_f32_16x16x32_bf16(false, a1, false, bi1, (short)0, accI11, false, false);
    }
  }

  // --- Epilogue: C/D layout: VGPR j -> M=j (lanes 0-15) / M=j+8 (lanes 16-31)
  #pragma unroll
  for (int ni = 0; ni < 2; ++ni) {
    const int n = n0 + wvN * 32 + ni * 16 + lr;
    const float brv   = b_r[n];
    const float biv   = b_i[n];
    const float decay = 8.0f * softplusf_(Lambda[n]);
    #pragma unroll
    for (int mi = 0; mi < 2; ++mi) {
      const int mb = m0 + wvM * 32 + mi * 16 + hi * 8;
      const v8f ar = (mi == 0) ? (ni == 0 ? accR00 : accR01)
                               : (ni == 0 ? accR10 : accR11);
      const v8f ai = (mi == 0) ? (ni == 0 ? accI00 : accI01)
                               : (ni == 0 ? accI10 : accI11);
      #pragma unroll
      for (int j = 0; j < 8; ++j) {
        const size_t idx = (size_t)(mb + j) * D + n;
        const float rg = sigmoidf_(ar[j] + brv);
        const float ig = sigmoidf_(ai[j] + biv);
        const float av = __expf(-decay * rg);
        const float sc = sqrtf(fmaxf(1.0f - av * av, 0.0f));
        a_out[idx]  = av;
        gx_out[idx] = sc * ig * x[idx];
      }
    }
  }
}

// ---------------------------------------------------------------------------
// Phase 2a: per-chunk local scan with h_in = 0; emit carry (h_end, prod a).
// One thread per (b, chunk, d); consecutive threads -> consecutive d (coalesced).
// ---------------------------------------------------------------------------
__global__ __launch_bounds__(256)
void rglru_scan_chunks(const float* __restrict__ a, const float* __restrict__ gx,
                       float* __restrict__ carry_h, float* __restrict__ carry_P)
{
  const int g = blockIdx.x * blockDim.x + threadIdx.x;   // [0, B*CH*D)
  const int d = g & (D - 1);
  const int c = (g >> 10) & (CH - 1);
  const int b = g >> 16;                                  // D*CH = 65536
  size_t base = (size_t)(b * T + c * LT) * D + d;
  float h = 0.0f, P = 1.0f;
  #pragma unroll 4
  for (int t = 0; t < LT; ++t) {
    const float av = a[base];
    const float gv = gx[base];
    h = fmaf(av, h, gv);
    P *= av;
    base += D;
  }
  carry_h[g] = h;
  carry_P[g] = P;
}

// ---------------------------------------------------------------------------
// Phase 2b: serial scan over the 64 chunk carries per (b,d); also emits h_T.
// ---------------------------------------------------------------------------
__global__ __launch_bounds__(256)
void rglru_scan_carries(const float* __restrict__ h0,
                        const float* __restrict__ carry_h,
                        const float* __restrict__ carry_P,
                        float* __restrict__ chunk_hin,
                        float* __restrict__ hT_out)
{
  const int g = blockIdx.x * blockDim.x + threadIdx.x;   // [0, B*D)
  const int d = g & (D - 1);
  const int b = g >> 10;
  float run = h0[g];
  for (int c = 0; c < CH; ++c) {
    const size_t i = (size_t)(b * CH + c) * D + d;
    chunk_hin[i] = run;                                   // exclusive prefix
    run = fmaf(carry_P[i], run, carry_h[i]);
  }
  hT_out[g] = run;
}

// ---------------------------------------------------------------------------
// Phase 2c: replay each chunk with its correct h_in, overwrite gx with h.
// ---------------------------------------------------------------------------
__global__ __launch_bounds__(256)
void rglru_scan_apply(const float* __restrict__ a,
                      const float* __restrict__ chunk_hin,
                      float* __restrict__ out /* holds gx; overwritten with h */)
{
  const int g = blockIdx.x * blockDim.x + threadIdx.x;   // [0, B*CH*D)
  const int d = g & (D - 1);
  const int c = (g >> 10) & (CH - 1);
  const int b = g >> 16;
  size_t base = (size_t)(b * T + c * LT) * D + d;
  float h = chunk_hin[g];
  #pragma unroll 4
  for (int t = 0; t < LT; ++t) {
    const float av = a[base];
    const float gv = out[base];
    h = fmaf(av, h, gv);
    out[base] = h;
    base += D;
  }
}

// ---------------------------------------------------------------------------
extern "C" void kernel_launch(void* const* d_in, const int* in_sizes, int n_in,
                              void* d_out, int out_size, void* d_ws, size_t ws_size,
                              hipStream_t stream)
{
  const float* x   = (const float*)d_in[0];
  const float* h0  = (const float*)d_in[1];
  const float* W_r = (const float*)d_in[2];
  const float* b_r = (const float*)d_in[3];
  const float* W_i = (const float*)d_in[4];
  const float* b_i = (const float*)d_in[5];
  const float* Lam = (const float*)d_in[6];

  float* out_seq = (float*)d_out;                 // (B,T,D)
  float* hT      = out_seq + (size_t)M * D;       // (B,D)

  // Workspace layout: a (64MB) | carry_h (1MB) | carry_P (1MB) | chunk_hin (1MB)
  float* ws_a      = (float*)d_ws;
  float* carry_h   = ws_a + (size_t)M * D;
  float* carry_P   = carry_h + (size_t)Bsz * CH * D;
  float* chunk_hin = carry_P + (size_t)Bsz * CH * D;

  dim3 gGemm(D / BN, M / BM);                     // (8, 256)
  rglru_gates_wmma<<<gGemm, 256, 0, stream>>>(x, W_r, b_r, W_i, b_i, Lam,
                                              ws_a, out_seq /* gx staged here */);
  rglru_scan_chunks<<<(Bsz * CH * D) / 256, 256, 0, stream>>>(ws_a, out_seq,
                                                              carry_h, carry_P);
  rglru_scan_carries<<<(Bsz * D) / 256, 256, 0, stream>>>(h0, carry_h, carry_P,
                                                          chunk_hin, hT);
  rglru_scan_apply<<<(Bsz * CH * D) / 256, 256, 0, stream>>>(ws_a, chunk_hin,
                                                             out_seq);
}
